// FastPointNet_28913719837076
// MI455X (gfx1250) — matrix operand order repaired and checked by reading, not compiled, not run on hardware
//
#include <hip/hip_runtime.h>

// ---------------------------------------------------------------------------
// Types for CDNA5 WMMA
// ---------------------------------------------------------------------------
typedef _Float16 half_t;
typedef __attribute__((ext_vector_type(16))) _Float16 v16h;
typedef __attribute__((ext_vector_type(8)))  float    v8f;

union FragU { uint4 u[2]; v16h h; };

#define LDSS 40   // LDS row stride in halves (80B: 16B-aligned, bank-conflict friendly)

static inline int pad32i(int v) { return (v + 31) & ~31; }

// ---------------------------------------------------------------------------
// Fused GEMM + bias + BN-statistics kernel.
//   C[M x ldc](f16) = A[M x K](f16) @ W[K x N] + bias   (W given n-major: Wt[n][k])
//   Also accumulates per-column sum / sum-of-squares (f32, from f32 accumulators).
// Block: 256 thr = 8 waves; tile 128(M) x 32(N); waves arranged 4x2, each 32x16.
// K is a multiple of 32 (inputs zero-padded); ldc = pad32(N) (pad cols stored 0).
// ---------------------------------------------------------------------------
__global__ __launch_bounds__(256) void gemm_wmma(
    const half_t* __restrict__ A,
    const half_t* __restrict__ Wt,
    const float*  __restrict__ bias,
    half_t* __restrict__ C,
    float* __restrict__ s_sum, float* __restrict__ s_ssq,
    int M, int K, int N, int ldc, int do_stats)
{
  __shared__ __align__(16) half_t As[128 * LDSS];
  __shared__ __align__(16) half_t Bs[32 * LDSS];

  const int tid  = threadIdx.x;
  const int lane = tid & 31;
  const int wave = tid >> 5;
  const int m0 = blockIdx.x * 128;
  const int n0 = blockIdx.y * 32;
  const int mw = (wave >> 1) * 32;   // wave's M offset within tile
  const int nw = (wave & 1) * 16;    // wave's N offset within tile
  const int l15  = lane & 15;
  const int hsel = lane >> 4;

  v8f acc0 = {0.f,0.f,0.f,0.f,0.f,0.f,0.f,0.f};
  v8f acc1 = {0.f,0.f,0.f,0.f,0.f,0.f,0.f,0.f};

  const int arow = tid >> 1;              // 0..127
  const int acol = (tid & 1) * 16;        // 0 or 16 halves
  const half_t* aBase = A + (size_t)(m0 + arow) * K + acol;
  const int brow = tid >> 1;              // n (0..31), tid<64 only
  const int bcol = (tid & 1) * 16;
  const half_t* bBase = Wt + (size_t)(n0 + brow) * K + bcol;

  for (int k0 = 0; k0 < K; k0 += 32) {
    __syncthreads();
    // A tile: 128 rows x 32 halves; each thread 16 halves (two b128 loads/stores)
    uint4 av0 = ((const uint4*)(aBase + k0))[0];
    uint4 av1 = ((const uint4*)(aBase + k0))[1];
    { uint4* d = (uint4*)(As + arow * LDSS + acol); d[0] = av0; d[1] = av1; }
    // B tile (n-major): 32 n-rows x 32 k-halves
    if (tid < 64) {
      uint4 bv0 = ((const uint4*)(bBase + k0))[0];
      uint4 bv1 = ((const uint4*)(bBase + k0))[1];
      uint4* d = (uint4*)(Bs + brow * LDSS + bcol); d[0] = bv0; d[1] = bv1;
    }
    __syncthreads();

    // B fragment: lane holds N = nw+l15, elements K = hsel*8 + 0..7 and 16+hsel*8+0..7
    FragU fb;
    { const half_t* p = Bs + (nw + l15) * LDSS + hsel * 8;
      fb.u[0] = *(const uint4*)p;
      fb.u[1] = *(const uint4*)(p + 16); }
    // A fragments: lane holds M = l15, elements K = hsel*8 + 0..7 and 16+hsel*8+0..7
    FragU fa0, fa1;
    { const half_t* p = As + (mw + l15) * LDSS + hsel * 8;
      fa0.u[0] = *(const uint4*)p;
      fa0.u[1] = *(const uint4*)(p + 16);
      const half_t* q = p + 16 * LDSS;
      fa1.u[0] = *(const uint4*)q;
      fa1.u[1] = *(const uint4*)(q + 16); }

    acc0 = __builtin_amdgcn_wmma_f32_16x16x32_f16(false, fa0.h, false, fb.h,
                                                  (short)0, acc0, false, false);
    acc1 = __builtin_amdgcn_wmma_f32_16x16x32_f16(false, fa1.h, false, fb.h,
                                                  (short)0, acc1, false, false);
  }

  // Epilogue: D layout: element v -> row = v + 8*hsel ; col = l15
  const int ncol = n0 + nw + l15;
  const float bv = bias[ncol];
  float s = 0.f, ss = 0.f;
#pragma unroll
  for (int sub = 0; sub < 2; ++sub) {
    v8f acc = sub ? acc1 : acc0;
    int rbase = m0 + mw + sub * 16 + hsel * 8;
#pragma unroll
    for (int v = 0; v < 8; ++v) {
      int r = rbase + v;
      if (r < M) {
        if (ncol < N) {
          float y = acc[v] + bv;
          C[(size_t)r * ldc + ncol] = (half_t)y;
          s += y; ss += y * y;
        } else {
          C[(size_t)r * ldc + ncol] = (half_t)0.f;   // keep pad columns zero
        }
      }
    }
  }
  if (do_stats && ncol < N) {
    s  += __shfl_xor(s, 16);
    ss += __shfl_xor(ss, 16);
    if (hsel == 0) {
      atomicAdd(&s_sum[ncol], s);
      atomicAdd(&s_ssq[ncol], ss);
    }
  }
}

// ---------------------------------------------------------------------------
// Small helper kernels
// ---------------------------------------------------------------------------
__global__ void zero_f32(float* p, int n) {
  int i = blockIdx.x * 256 + threadIdx.x;
  if (i < n) p[i] = 0.f;
}

// f32 [K x N] row-major  ->  f16 [Npad x Kpad] n-major, zero padded
__global__ void prep_weight(const float* __restrict__ W, half_t* __restrict__ Wt,
                            int K, int N, int Kp, int Np) {
  int i = blockIdx.x * 256 + threadIdx.x;
  int tot = Np * Kp;
  if (i >= tot) return;
  int n = i / Kp, k = i % Kp;
  float v = (k < K && n < N) ? W[(size_t)k * N + n] : 0.f;
  Wt[i] = (half_t)v;
}

__global__ void pad_vec(const float* __restrict__ s, float* __restrict__ d, int N, int Np) {
  int i = blockIdx.x * 256 + threadIdx.x;
  if (i < Np) d[i] = (i < N) ? s[i] : 0.f;
}

__global__ void sample_pos(const float* __restrict__ p, const int* __restrict__ idx,
                           float* __restrict__ o, int n) {
  int i = blockIdx.x * 256 + threadIdx.x;
  if (i >= n) return;
  int s = idx[i];
  o[i * 3 + 0] = p[(size_t)s * 3 + 0];
  o[i * 3 + 1] = p[(size_t)s * 3 + 1];
  o[i * 3 + 2] = p[(size_t)s * 3 + 2];
}

// Build SA message matrix: [xprev[psrc] | posf[psrc]-posc[r>>5] | 0-pad] -> f16
__global__ void gather_sa(const float* __restrict__ xprev, int Cin,
                          const float* __restrict__ posf, const float* __restrict__ posc,
                          const int* __restrict__ psrc, half_t* __restrict__ out,
                          int rows, int Kp) {
  long long i = (long long)blockIdx.x * 256 + threadIdx.x;
  long long tot = (long long)rows * Kp;
  if (i >= tot) return;
  int r = (int)(i / Kp), ch = (int)(i % Kp);
  float v = 0.f;
  if (ch < Cin + 3) {
    int src = psrc[r];
    if (ch < Cin) v = xprev[(size_t)src * Cin + ch];
    else {
      int tgt = r >> 5;               // KP = 32, ptgt = repeat(arange, 32)
      int j = ch - Cin;
      v = posf[(size_t)src * 3 + j] - posc[(size_t)tgt * 3 + j];
    }
  }
  out[i] = (half_t)v;
}

// Inverse-distance interpolation (KI=3 contiguous) + skip-concat -> padded f16
__global__ void interp_concat(const float* __restrict__ xc, int Cc,
                              const float* __restrict__ posc, const float* __restrict__ posf,
                              const int* __restrict__ isrc,
                              const float* __restrict__ xskip, int Cs,
                              half_t* __restrict__ out, int nf, int Kp) {
  long long i = (long long)blockIdx.x * 256 + threadIdx.x;
  long long tot = (long long)nf * Kp;
  if (i >= tot) return;
  int f = (int)(i / Kp), ch = (int)(i % Kp);
  float v = 0.f;
  if (ch < Cc) {
    float px = posf[(size_t)f * 3 + 0];
    float py = posf[(size_t)f * 3 + 1];
    float pz = posf[(size_t)f * 3 + 2];
    float num = 0.f, den = 0.f;
#pragma unroll
    for (int j = 0; j < 3; ++j) {
      int c = isrc[(size_t)f * 3 + j];
      float dx = posc[(size_t)c * 3 + 0] - px;
      float dy = posc[(size_t)c * 3 + 1] - py;
      float dz = posc[(size_t)c * 3 + 2] - pz;
      float d2 = fmaxf(dx * dx + dy * dy + dz * dz, 1e-16f);
      float w = 1.f / d2;
      num += xc[(size_t)c * Cc + ch] * w;
      den += w;
    }
    v = num / den;
  } else if (ch < Cc + Cs) {
    v = xskip[(size_t)f * Cs + (ch - Cc)];
  }
  out[i] = (half_t)v;
}

__global__ void bn_finalize(const float* __restrict__ s_sum, const float* __restrict__ s_ssq,
                            const float* __restrict__ g, const float* __restrict__ bt,
                            float* __restrict__ scale, float* __restrict__ shift,
                            int N, float invM) {
  int c = blockIdx.x * 128 + threadIdx.x;
  if (c >= N) return;
  float mu  = s_sum[c] * invM;
  float var = s_ssq[c] * invM - mu * mu;
  float inv = rsqrtf(var + 1e-5f);
  float sc  = g[c] * inv;
  scale[c] = sc;
  shift[c] = bt[c] - mu * sc;
}

__global__ void bn_relu(half_t* __restrict__ C, const float* __restrict__ scale,
                        const float* __restrict__ shift, int M, int N, int ldc) {
  long long i = (long long)blockIdx.x * 256 + threadIdx.x;
  long long tot = (long long)M * N;
  if (i >= tot) return;
  int r = (int)(i / N), c = (int)(i % N);
  size_t idx = (size_t)r * ldc + c;
  float y = (float)C[idx] * scale[c] + shift[c];
  C[idx] = (half_t)fmaxf(y, 0.f);
}

// segment_max over 32 contiguous rows -> f32 [nc x C]
__global__ void seg_max32(const half_t* __restrict__ msg, int ldm,
                          float* __restrict__ out, int nc, int C) {
  long long i = (long long)blockIdx.x * 256 + threadIdx.x;
  long long tot = (long long)nc * C;
  if (i >= tot) return;
  int t = (int)(i / C), ch = (int)(i % C);
  const half_t* p = msg + (size_t)t * 32 * ldm + ch;
  float m = -3.0e38f;
#pragma unroll 4
  for (int j = 0; j < 32; ++j) m = fmaxf(m, (float)p[(size_t)j * ldm]);
  out[i] = m;
}

__global__ void act_to_f32(const half_t* __restrict__ act, int ld,
                           float* __restrict__ out, int rows, int C) {
  long long i = (long long)blockIdx.x * 256 + threadIdx.x;
  long long tot = (long long)rows * C;
  if (i >= tot) return;
  int r = (int)(i / C), c = (int)(i % C);
  out[i] = (float)act[(size_t)r * ld + c];
}

// final plain Linear 128 -> 3 (reads raw f32 weights)
__global__ void out_linear(const float* __restrict__ xin, const float* __restrict__ W,
                           const float* __restrict__ b, float* __restrict__ out, int P) {
  int p = blockIdx.x * 256 + threadIdx.x;
  if (p >= P) return;
  float o0 = b[0], o1 = b[1], o2 = b[2];
  const float* xr = xin + (size_t)p * 128;
#pragma unroll 8
  for (int k = 0; k < 128; ++k) {
    float x = xr[k];
    o0 += x * W[k * 3 + 0];
    o1 += x * W[k * 3 + 1];
    o2 += x * W[k * 3 + 2];
  }
  out[p * 3 + 0] = o0; out[p * 3 + 1] = o1; out[p * 3 + 2] = o2;
}

// ---------------------------------------------------------------------------
// Host orchestration
// ---------------------------------------------------------------------------
extern "C" void kernel_launch(void* const* d_in, const int* in_sizes, int n_in,
                              void* d_out, int out_size, void* d_ws, size_t ws_size,
                              hipStream_t stream) {
  (void)in_sizes; (void)n_in; (void)out_size; (void)ws_size;

  static const int SZ[6] = {100000, 25000, 6250, 1600, 400, 100};
  static const int XC_C[6] = {3, 64, 128, 212, 256, 256};  // channels of xc at each level

  const float* x   = (const float*)d_in[0];
  const float* pos = (const float*)d_in[1];
  const int* sample[5]; const int* psrc[5]; const int* isrc[5];
  for (int s = 0; s < 5; ++s) {
    sample[s] = (const int*)d_in[2 + 5 * s];
    psrc[s]   = (const int*)d_in[3 + 5 * s];
    isrc[s]   = (const int*)d_in[5 + 5 * s];
  }

  // params leaves assumed in jax tree order: sorted MLP names
  //   fp0,fp1,fp2,fp3,fp4,out,sa0,sa1,sa2,sa3,sa4
  // and per BN-layer sorted keys: W, b, bt, g  (plain 'out' layer: W, b).
  struct MlpP {
    int nl; int dims[5];
    const float *W[4], *b[4], *g[4], *bt[4];
    half_t* Wt[4]; float* bp[4];
  };
  static const int ND[11][6] = {
    {4, 131, 128, 128, 128, 0},   // fp0
    {3, 276, 212, 128,   0, 0},   // fp1
    {3, 340, 212, 212,   0, 0},   // fp2
    {3, 468, 212, 212,   0, 0},   // fp3
    {3, 512, 256, 256,   0, 0},   // fp4
    {2, 128,   3,   0,   0, 0},   // out (plain)
    {4,   6,  32,  32,  64, 0},   // sa0
    {4,  67,  64,  64, 128, 0},   // sa1
    {4, 131, 128, 128, 212, 0},   // sa2
    {4, 215, 212, 212, 256, 0},   // sa3
    {4, 259, 256, 256, 256, 0},   // sa4
  };
  MlpP mp[11];
  int pi = 27;
  for (int i = 0; i < 11; ++i) {
    int nd = ND[i][0];
    mp[i].nl = nd - 1;
    for (int j = 0; j < nd; ++j) mp[i].dims[j] = ND[i][1 + j];
    bool plain = (i == 5);
    for (int l = 0; l < mp[i].nl; ++l) {
      mp[i].W[l] = (const float*)d_in[pi++];
      mp[i].b[l] = (const float*)d_in[pi++];
      if (!plain) {
        mp[i].bt[l] = (const float*)d_in[pi++];
        mp[i].g[l]  = (const float*)d_in[pi++];
      } else { mp[i].bt[l] = nullptr; mp[i].g[l] = nullptr; }
    }
  }

  // ---- workspace bump allocator
  size_t off = 0;
  auto alloc = [&](size_t bytes) -> void* {
    off = (off + 255) & ~(size_t)255;
    void* p = (char*)d_ws + off;
    off += bytes;
    return p;
  };
  const size_t ACT_CAP = 51250000;  // halves; covers 800000 x 64 (sa0 widest stage)
  half_t* actA = (half_t*)alloc(ACT_CAP * sizeof(half_t));
  half_t* actB = (half_t*)alloc(ACT_CAP * sizeof(half_t));
  float* xcL[6]; xcL[0] = nullptr;
  for (int lv = 1; lv <= 5; ++lv)
    xcL[lv] = (float*)alloc((size_t)SZ[lv] * XC_C[lv] * sizeof(float));
  float* fpOutA = (float*)alloc((size_t)100000 * 128 * sizeof(float)); // steps 4,2,0
  float* fpOutB = (float*)alloc((size_t)25000 * 128 * sizeof(float));  // steps 3,1
  float* posL[6]; posL[0] = nullptr;
  for (int lv = 1; lv <= 5; ++lv)
    posL[lv] = (float*)alloc((size_t)SZ[lv] * 3 * sizeof(float));
  float* stats  = (float*)alloc(1024 * sizeof(float));
  float* scaleB = (float*)alloc(512 * sizeof(float));
  float* shiftB = (float*)alloc(512 * sizeof(float));

  // ---- weight preparation (f32 [KxN] -> f16 n-major [Npad x Kpad], padded bias)
  for (int i = 0; i < 11; ++i) {
    if (i == 5) continue;  // 'out' uses raw f32 weights
    for (int l = 0; l < mp[i].nl; ++l) {
      int K = mp[i].dims[l], N = mp[i].dims[l + 1];
      int Kp = pad32i(K), Np = pad32i(N);
      mp[i].Wt[l] = (half_t*)alloc((size_t)Np * Kp * sizeof(half_t));
      mp[i].bp[l] = (float*)alloc((size_t)Np * sizeof(float));
      int tot = Np * Kp;
      prep_weight<<<(tot + 255) / 256, 256, 0, stream>>>(mp[i].W[l], mp[i].Wt[l], K, N, Kp, Np);
      pad_vec<<<(Np + 255) / 256, 256, 0, stream>>>(mp[i].b[l], mp[i].bp[l], N, Np);
    }
  }

  // ---- hierarchical position sampling: posL[s+1] = posL[s][sample[s]]
  {
    const float* pprev = pos;
    for (int s = 0; s < 5; ++s) {
      int n = SZ[s + 1];
      sample_pos<<<(n + 255) / 256, 256, 0, stream>>>(pprev, sample[s], posL[s + 1], n);
      pprev = posL[s + 1];
    }
  }

  // ---- MLP runner (Linear -> BN(batch stats) -> ReLU per layer), all BN layers
  auto run_mlp = [&](MlpP& m, int rows) -> half_t* {
    half_t* bin = actA; half_t* bout = actB;
    for (int l = 0; l < m.nl; ++l) {
      int K = m.dims[l], N = m.dims[l + 1];
      int Kp = pad32i(K), Np = pad32i(N);
      zero_f32<<<(2 * Np + 255) / 256, 256, 0, stream>>>(stats, 2 * Np);
      dim3 grid((rows + 127) / 128, Np / 32);
      gemm_wmma<<<grid, 256, 0, stream>>>(bin, m.Wt[l], m.bp[l], bout,
                                          stats, stats + Np, rows, Kp, N, Np, 1);
      bn_finalize<<<(N + 127) / 128, 128, 0, stream>>>(stats, stats + Np, m.g[l], m.bt[l],
                                                       scaleB, shiftB, N, 1.0f / (float)rows);
      long long tot = (long long)rows * N;
      bn_relu<<<(unsigned)((tot + 255) / 256), 256, 0, stream>>>(bout, scaleB, shiftB, rows, N, Np);
      half_t* t = bin; bin = bout; bout = t;
    }
    return bin;  // final layer output
  };

  // ---- Set abstraction: gather -> MLP -> segment-max (KP=32 contiguous)
  for (int s = 0; s < 5; ++s) {
    MlpP& m = mp[6 + s];
    int nc = SZ[s + 1];
    int rows = nc * 32;
    int Cin = XC_C[s];
    int Kp0 = pad32i(m.dims[0]);
    const float* xprev = (s == 0) ? x : xcL[s];
    const float* pf = (s == 0) ? pos : posL[s];
    long long tot = (long long)rows * Kp0;
    gather_sa<<<(unsigned)((tot + 255) / 256), 256, 0, stream>>>(
        xprev, Cin, pf, posL[s + 1], psrc[s], actA, rows, Kp0);
    half_t* res = run_mlp(m, rows);
    int Cout = m.dims[m.nl];
    int ldr = pad32i(Cout);
    long long tot2 = (long long)nc * Cout;
    seg_max32<<<(unsigned)((tot2 + 255) / 256), 256, 0, stream>>>(res, ldr, xcL[s + 1], nc, Cout);
  }

  // ---- Feature propagation: interp(KI=3) + skip concat -> MLP
  const float* coarse_xc = xcL[5];
  int coarse_C = 256;
  for (int s = 4; s >= 0; --s) {
    MlpP& m = mp[s];  // fp_s
    int nf = SZ[s];
    int Cs = XC_C[s];
    const float* xskip = (s == 0) ? x : xcL[s];
    const float* pf = (s == 0) ? pos : posL[s];
    const float* pc = posL[s + 1];
    int Kp0 = pad32i(m.dims[0]);
    long long tot = (long long)nf * Kp0;
    interp_concat<<<(unsigned)((tot + 255) / 256), 256, 0, stream>>>(
        coarse_xc, coarse_C, pc, pf, isrc[s], xskip, Cs, actA, nf, Kp0);
    half_t* res = run_mlp(m, nf);
    int Cout = m.dims[m.nl];
    int ldr = pad32i(Cout);
    float* dst = (s % 2 == 0) ? fpOutA : fpOutB;
    long long tot2 = (long long)nf * Cout;
    act_to_f32<<<(unsigned)((tot2 + 255) / 256), 256, 0, stream>>>(res, ldr, dst, nf, Cout);
    coarse_xc = dst;
    coarse_C = Cout;
  }

  // ---- output head: plain Linear [128 -> 3]
  out_linear<<<(100000 + 255) / 256, 256, 0, stream>>>(
      fpOutA, mp[5].W[0], mp[5].b[0], (float*)d_out, 100000);
}
